// TransImg_90658169684632
// MI455X (gfx1250) — matrix operand order repaired and checked by reading, not compiled, not run on hardware
//
#include <hip/hip_runtime.h>
#include <math.h>

#define NNODE 10000
#define NEDGE 60000
#define TBN 128          // block tile: 64 rows x 128 cols, 8 waves, 4 wmma/wave/chunk
#define LDP 40           // padded LDS row stride (halves): 80B, 16B-aligned

typedef __attribute__((ext_vector_type(16))) __bf16 v16bf;
typedef __attribute__((ext_vector_type(8)))  __bf16 v8bf;
typedef __attribute__((ext_vector_type(2)))  __bf16 bf16x2;
typedef __attribute__((ext_vector_type(8)))  float  v8f;

typedef __attribute__((address_space(3))) unsigned char lds_byte;

__device__ __forceinline__ lds_byte* lds_ptr(const void* p)
{
  return (lds_byte*)(void*)p;   // addrspacecast generic -> LDS (32-bit offset)
}

// One async 16B copy: LDS[lds + inst_off] = MEM[gaddr + inst_off]; ASYNCcnt++
#define ASYNC_B128(LDS, GA)                                                   \
  asm volatile("global_load_async_to_lds_b128 %0, %1, off"                    \
               :: "v"(LDS), "v"(GA) : "memory")
#define ASYNC_B128_OFF16(LDS, GA)                                             \
  asm volatile("global_load_async_to_lds_b128 %0, %1, off offset:16"          \
               :: "v"(LDS), "v"(GA) : "memory")

__device__ __forceinline__
v16bf cat8(v8bf lo, v8bf hi)
{
  return __builtin_shufflevector(lo, hi, 0,1,2,3,4,5,6,7,8,9,10,11,12,13,14,15);
}

// ---------------------------------------------------------------------------
// Layout pre-conversion (once per layer / per weight matrix):
//   Xb[Mp][Kp]  bf16 row-major, zero-padded  (Mp = ceil64(M), Kp = ceil64(K))
//   Wf[Ndp][Kp] bf16 "fragment order": column n -> contiguous K halves
// Kp multiple of 64 -> even chunk count -> branch-free 2x-unrolled pipeline.
// ---------------------------------------------------------------------------
__global__ __launch_bounds__(256)
void cvt_x_bf16(const float* __restrict__ X, __bf16* __restrict__ Xb,
                int M, int K, int Kp)
{
  const int r = blockIdx.y;
  const int c = (blockIdx.x * blockDim.x + threadIdx.x) << 1;
  if (c >= Kp) return;
  float x0 = 0.f, x1 = 0.f;
  if (r < M) {
    const float* row = X + (size_t)r * K;
    if (c < K)     x0 = row[c];
    if (c + 1 < K) x1 = row[c + 1];
  }
  bf16x2 p; p.x = (__bf16)x0; p.y = (__bf16)x1;
  *(bf16x2*)&Xb[(size_t)r * Kp + c] = p;
}

__global__ __launch_bounds__(256)
void cvt_w_bf16(const float* __restrict__ W, __bf16* __restrict__ Wf,
                int K, int Nd, int Kp)
{
  const int n = blockIdx.y;
  const int c = (blockIdx.x * blockDim.x + threadIdx.x) << 1;
  if (c >= Kp) return;
  float w0 = 0.f, w1 = 0.f;
  if (n < Nd) {
    if (c < K)     w0 = W[(size_t)c * Nd + n];
    if (c + 1 < K) w1 = W[(size_t)(c + 1) * Nd + n];
  }
  bf16x2 p; p.x = (__bf16)w0; p.y = (__bf16)w1;
  *(bf16x2*)&Wf[(size_t)n * Kp + c] = p;
}

// ---------------------------------------------------------------------------
// One K-chunk of WMMAs: all 10 ds_load_b128 issued before 4 back-to-back wmmas.
// ---------------------------------------------------------------------------
__device__ __forceinline__
void compute_chunk(const __bf16 (*Xsb)[LDP], const __bf16 (*Wtb)[LDP],
                   int wave, int l15, int lhalf, v8f acc[4])
{
  const __bf16* bcol = &Wtb[(wave << 4) + l15][0];
  const v16bf b = cat8(*(const v8bf*)(bcol + (lhalf << 3)),
                       *(const v8bf*)(bcol + 16 + (lhalf << 3)));
  v16bf a[4];
#pragma unroll
  for (int s = 0; s < 4; ++s) {
    const __bf16* arow = &Xsb[(s << 4) + l15][0];
    a[s] = cat8(*(const v8bf*)(arow + (lhalf << 3)),
                *(const v8bf*)(arow + 16 + (lhalf << 3)));
  }
#pragma unroll
  for (int s = 0; s < 4; ++s)
    acc[s] = __builtin_amdgcn_wmma_f32_16x16x32_bf16(false, a[s], false, b,
                                                     (short)0, acc[s],
                                                     false, false);
}

// ---------------------------------------------------------------------------
// C[M,Nd] = Xb[M,Kp] @ Wf[Nd,Kp]^T(frag) + bias  (bf16 WMMA, f32 accumulate)
// Block: 256 thr = 8 waves; tile 64x128; wave computes 64x16 (4 wmma/chunk).
// Double-buffered LDS staged via global_load_async_to_lds_b128; pipeline
// unrolled 2x (nch guaranteed even) -> compile-time buffer indices.
// ---------------------------------------------------------------------------
__global__ __launch_bounds__(256)
void gemm_bias_wmma(const __bf16* __restrict__ Xb, const __bf16* __restrict__ Wf,
                    const float* __restrict__ bias, float* __restrict__ C,
                    int M, int Kp, int Nd)
{
  __shared__ __bf16 Xs[2][64][LDP];
  __shared__ __bf16 Wt[2][TBN][LDP];

  const int tid   = threadIdx.x;
  const int wave  = tid >> 5;
  const int lane  = tid & 31;
  const int l15   = lane & 15;
  const int lhalf = lane >> 4;
  const int m0 = blockIdx.x << 6;
  const int nb = blockIdx.y * TBN;

  // per-thread async-copy assignments (16B granules)
  const __bf16* xg = Xb + (size_t)(m0 + (tid >> 2)) * Kp + ((tid & 3) << 3);
  const __bf16* wg = Wf + (size_t)(nb + (tid >> 1)) * Kp + ((tid & 1) << 4);
  lds_byte* lx0 = lds_ptr(&Xs[0][tid >> 2][(tid & 3) << 3]);
  lds_byte* lx1 = lds_ptr(&Xs[1][tid >> 2][(tid & 3) << 3]);
  lds_byte* lw0 = lds_ptr(&Wt[0][tid >> 1][(tid & 1) << 4]);
  lds_byte* lw1 = lds_ptr(&Wt[1][tid >> 1][(tid & 1) << 4]);

  v8f acc[4] = {v8f{}, v8f{}, v8f{}, v8f{}};
  const int nch = Kp >> 5;            // even (Kp multiple of 64)

  // prologue: chunk 0 -> buffer 0
  ASYNC_B128(lx0, xg);
  ASYNC_B128(lw0, wg);
  ASYNC_B128_OFF16(lw0, wg);
  xg += 32; wg += 32;

  for (int c = 0; c < nch; c += 2) {
    // ---- even chunk: issue c+1 -> buf1, compute buf0 ----
    ASYNC_B128(lx1, xg);
    ASYNC_B128(lw1, wg);
    ASYNC_B128_OFF16(lw1, wg);
    xg += 32; wg += 32;
    asm volatile("s_wait_asynccnt 0x3" ::: "memory");   // chunk c landed (in-order)
    __syncthreads();
    compute_chunk(Xs[0], Wt[0], wave, l15, lhalf, acc);
    __syncthreads();                                    // buf0 free for reuse

    // ---- odd chunk: issue c+2 -> buf0 (if any), compute buf1 ----
    if (c + 2 < nch) {
      ASYNC_B128(lx0, xg);
      ASYNC_B128(lw0, wg);
      ASYNC_B128_OFF16(lw0, wg);
      xg += 32; wg += 32;
      asm volatile("s_wait_asynccnt 0x3" ::: "memory"); // chunk c+1 landed
    } else {
      asm volatile("s_wait_asynccnt 0x0" ::: "memory");
    }
    __syncthreads();
    compute_chunk(Xs[1], Wt[1], wave, l15, lhalf, acc);
    __syncthreads();                                    // buf1 free for reuse
  }

  const int n = nb + (wave << 4) + l15;
  if (n < Nd) {
    const float bv = bias[n];
#pragma unroll
    for (int s = 0; s < 4; ++s) {
#pragma unroll
      for (int i = 0; i < 8; ++i) {    // C/D: VGPR i -> M = i (+8 for hi half)
        const int m = m0 + (s << 4) + i + (lhalf << 3);
        if (m < M) C[(size_t)m * Nd + n] = acc[s][i] + bv;
      }
    }
  }
}

// ---------------------------------------------------------------------------
__global__ void zero_u32(unsigned* __restrict__ p, int n)
{
  int i = blockIdx.x * blockDim.x + threadIdx.x;
  if (i < n) p[i] = 0u;
}

// One wave32 per edge: logits[e] = scale * dot(q[dst], k[src]); segment max
// via order-preserving uint encoding + atomicMax.
__global__ __launch_bounds__(256)
void edge_logits_max(const float* __restrict__ q, const float* __restrict__ kmat,
                     const int* __restrict__ src, const int* __restrict__ dst,
                     float* __restrict__ logits, unsigned* __restrict__ menc,
                     int E, int D, float scale)
{
  int e = (blockIdx.x << 3) + (threadIdx.x >> 5);
  if (e >= E) return;
  int lane = threadIdx.x & 31;
  const float* qp = q    + (size_t)dst[e] * D;
  const float* kp = kmat + (size_t)src[e] * D;
  float sum = 0.f;
  for (int d = lane; d < D; d += 32) sum += qp[d] * kp[d];
#pragma unroll
  for (int off = 16; off > 0; off >>= 1) sum += __shfl_down(sum, off, 32);
  if (lane == 0) {
    float l = sum * scale;
    logits[e] = l;
    unsigned u   = __float_as_uint(l);
    unsigned enc = (u & 0x80000000u) ? ~u : (u | 0x80000000u);
    atomicMax(menc + dst[e], enc);
  }
}

__global__ void edge_expsum(const float* __restrict__ logits,
                            const int* __restrict__ dst,
                            const unsigned* __restrict__ menc,
                            float* __restrict__ ebuf, float* __restrict__ ssum,
                            int E)
{
  int e = blockIdx.x * blockDim.x + threadIdx.x;
  if (e >= E) return;
  unsigned u = menc[dst[e]];
  float m = (u & 0x80000000u) ? __uint_as_float(u & 0x7fffffffu)
                              : __uint_as_float(~u);
  float x = expf(logits[e] - m);
  ebuf[e] = x;
  atomicAdd(ssum + dst[e], x);
}

__global__ __launch_bounds__(256)
void edge_agg(const float* __restrict__ ebuf, const float* __restrict__ ssum,
              const float* __restrict__ v, const int* __restrict__ src,
              const int* __restrict__ dst, float* __restrict__ out, int E, int D)
{
  int e = blockIdx.x;
  int d = blockIdx.y * blockDim.x + threadIdx.x;
  if (d >= D) return;
  int t = dst[e];
  float alpha = ebuf[e] / (ssum[t] + 1e-16f);
  atomicAdd(out + (size_t)t * D + d, alpha * v[(size_t)src[e] * D + d]);
}

__global__ void elu_kernel(float* __restrict__ x, size_t n)
{
  size_t i = (size_t)blockIdx.x * blockDim.x + threadIdx.x;
  if (i < n) { float v = x[i]; x[i] = (v > 0.f) ? v : expm1f(v); }
}

__global__ void concat2(const float* __restrict__ a, const float* __restrict__ b,
                        float* __restrict__ out, int nrow, int da, int db)
{
  int i = blockIdx.x * blockDim.x + threadIdx.x;
  int D = da + db;
  if (i >= nrow * D) return;
  int r = i / D, c = i - r * D;
  out[i] = (c < da) ? a[(size_t)r * da + c] : b[(size_t)r * db + (c - da)];
}

// ---------------------------------------------------------------------------
extern "C" void kernel_launch(void* const* d_in, const int* in_sizes, int n_in,
                              void* d_out, int out_size, void* d_ws, size_t ws_size,
                              hipStream_t stream)
{
  (void)in_sizes; (void)n_in; (void)out_size; (void)ws_size;
  const int Nn = NNODE, Ee = NEDGE;
  const int Mp = (Nn + 63) & ~63;          // 10048

  const float* features = (const float*)d_in[0];
  const float* imgf     = (const float*)d_in[1];
  const int*   ei       = (const int*)d_in[2];
  const int*   src      = ei;
  const int*   dst      = ei + Ee;
  auto cp = [&](int layer, int j) -> const float* {
    return (const float*)d_in[3 + layer * 8 + j];   // Wq,bq,Wk,bk,Wv,bv,Ws,bs
  };

  char* base = (char*)d_ws;
  size_t off = 0;
  auto carve = [&](size_t bytes) -> void* {
    void* p = base + off;
    off += (bytes + 255) & ~(size_t)255;
    return p;
  };
  float*    qb     = (float*)carve((size_t)Nn * 3000 * 4);
  float*    kb     = (float*)carve((size_t)Nn * 3000 * 4);
  float*    vb     = (float*)carve((size_t)Nn * 3000 * 4);
  float*    h1     = (float*)carve((size_t)Nn * 512 * 4);
  float*    h3     = (float*)carve((size_t)Nn * 512 * 4);
  float*    i1b    = (float*)carve((size_t)Nn * 512 * 4);
  float*    i3b    = (float*)carve((size_t)Nn * 512 * 4);
  float*    c3b    = (float*)carve((size_t)Nn * 512 * 4);
  float*    ccat   = (float*)carve((size_t)Nn * 128 * 4);
  float*    comb   = (float*)carve((size_t)Nn * 64 * 4);
  float*    logits = (float*)carve((size_t)Ee * 4);
  float*    ebuf   = (float*)carve((size_t)Ee * 4);
  unsigned* menc   = (unsigned*)carve((size_t)Nn * 4 * 2); // menc + ssum
  float*    ssum   = (float*)(menc + Nn);
  __bf16*   Xbf    = (__bf16*)carve((size_t)Mp * 3008 * 2);   // bf16 X, padded
  __bf16*   Wff    = (__bf16*)carve((size_t)3072 * 3008 * 2); // bf16 W, frag order

  auto cvtX = [&](const float* X, int K, int Kp) {
    dim3 g(((Kp >> 1) + 255) / 256, Mp);
    cvt_x_bf16<<<g, 256, 0, stream>>>(X, Xbf, Nn, K, Kp);
  };
  auto cvtW = [&](const float* W, int K, int Nd, int Kp, int Ndp) {
    dim3 g(((Kp >> 1) + 255) / 256, Ndp);
    cvt_w_bf16<<<g, 256, 0, stream>>>(W, Wff, K, Nd, Kp);
  };

  auto tconv = [&](const float* x, int din, int dout, int pl, float* out) {
    const int Kp  = (din + 63) & ~63;          // multiple of 64 -> even chunks
    const int Ndp = (dout + TBN - 1) & ~(TBN - 1);
    dim3 grid(Mp / 64, Ndp / TBN);
    cvtX(x, din, Kp);
    cvtW(cp(pl, 0), din, dout, Kp, Ndp);
    gemm_bias_wmma<<<grid, 256, 0, stream>>>(Xbf, Wff, cp(pl, 1), qb, Nn, Kp, dout);
    cvtW(cp(pl, 2), din, dout, Kp, Ndp);
    gemm_bias_wmma<<<grid, 256, 0, stream>>>(Xbf, Wff, cp(pl, 3), kb, Nn, Kp, dout);
    cvtW(cp(pl, 4), din, dout, Kp, Ndp);
    gemm_bias_wmma<<<grid, 256, 0, stream>>>(Xbf, Wff, cp(pl, 5), vb, Nn, Kp, dout);
    cvtW(cp(pl, 6), din, dout, Kp, Ndp);
    gemm_bias_wmma<<<grid, 256, 0, stream>>>(Xbf, Wff, cp(pl, 7), out, Nn, Kp, dout);

    zero_u32<<<(2 * Nn + 255) / 256, 256, 0, stream>>>(menc, 2 * Nn);
    float scale = 1.f / sqrtf((float)dout);
    edge_logits_max<<<(Ee + 7) / 8, 256, 0, stream>>>(qb, kb, src, dst,
                                                      logits, menc, Ee, dout, scale);
    edge_expsum<<<(Ee + 255) / 256, 256, 0, stream>>>(logits, dst, menc,
                                                      ebuf, ssum, Ee);
    edge_agg<<<dim3(Ee, (dout + 255) / 256), 256, 0, stream>>>(ebuf, ssum, vb,
                                                               src, dst, out, Ee, dout);
  };
  auto elu = [&](float* x, size_t n) {
    elu_kernel<<<(unsigned)((n + 255) / 256), 256, 0, stream>>>(x, n);
  };

  float* out_h2 = (float*)d_out;
  float* out_i2 = out_h2 + (size_t)Nn * 64;
  float* out_c2 = out_i2 + (size_t)Nn * 64;
  float* out_h4 = out_c2 + (size_t)Nn * 64;
  float* out_i4 = out_h4 + (size_t)Nn * 3000;
  float* out_c4 = out_i4 + (size_t)Nn * 3000;

  // node-feature branch
  tconv(features, 3000, 512, 0, h1);  elu(h1, (size_t)Nn * 512);
  tconv(h1, 512, 64, 1, out_h2);
  tconv(out_h2, 64, 512, 2, h3);      elu(h3, (size_t)Nn * 512);
  tconv(h3, 512, 3000, 3, out_h4);

  // image branch
  tconv(imgf, 1024, 512, 4, i1b);     elu(i1b, (size_t)Nn * 512);
  tconv(i1b, 512, 64, 5, out_i2);
  tconv(out_i2, 64, 512, 6, i3b);     elu(i3b, (size_t)Nn * 512);
  tconv(i3b, 512, 3000, 7, out_i4);

  // neck
  concat2<<<(Nn * 128 + 255) / 256, 256, 0, stream>>>(out_h2, out_i2, ccat, Nn, 64, 64);
  tconv(ccat, 128, 64, 8, comb);      elu(comb, (size_t)Nn * 64);
  tconv(comb, 64, 64, 9, out_c2);
  tconv(out_c2, 64, 512, 10, c3b);    elu(c3b, (size_t)Nn * 512);
  tconv(c3b, 512, 3000, 11, out_c4);
}
